// BasisAttention_49667001811349
// MI455X (gfx1250) — matrix operand order repaired and checked by reading, not compiled, not run on hardware
//
#include <hip/hip_runtime.h>
#include <hip/hip_bf16.h>
#include <math.h>

typedef __attribute__((ext_vector_type(16))) __bf16 v16bf;
typedef __attribute__((ext_vector_type(8)))  __bf16 v8bf;
typedef __attribute__((ext_vector_type(8)))  float  v8f;

#define HID   1024
#define NEMB  4096
#define KTOP  64
#define BATCH 16
#define SEQ   2048
#define TOK   (BATCH * SEQ)
#define EPSV  1e-6f

// ---------------------------------------------------------------------------
// Kernel 1: row-wise RMSNorm (D=1024), write bf16 (and optionally f32)
// ---------------------------------------------------------------------------
__launch_bounds__(256)
__global__ void rmsnorm_rows_kernel(const float* __restrict__ X,
                                    const float* __restrict__ w,
                                    __bf16* __restrict__ out_bf,
                                    float* __restrict__ out_f32) {
  const int row = blockIdx.x;
  const int tid = threadIdx.x;
  const float* xr = X + (size_t)row * HID;
  float v0 = xr[tid], v1 = xr[tid + 256], v2 = xr[tid + 512], v3 = xr[tid + 768];
  __shared__ float red[256];
  red[tid] = v0 * v0 + v1 * v1 + v2 * v2 + v3 * v3;
  __syncthreads();
  for (int s = 128; s > 0; s >>= 1) {
    if (tid < s) red[tid] += red[tid + s];
    __syncthreads();
  }
  const float r = rsqrtf(red[0] * (1.0f / HID) + EPSV);
  const float w0 = w[tid], w1 = w[tid + 256], w2 = w[tid + 512], w3 = w[tid + 768];
  const float o0 = v0 * r * w0, o1 = v1 * r * w1, o2 = v2 * r * w2, o3 = v3 * r * w3;
  __bf16* ob = out_bf + (size_t)row * HID;
  ob[tid] = (__bf16)o0; ob[tid + 256] = (__bf16)o1;
  ob[tid + 512] = (__bf16)o2; ob[tid + 768] = (__bf16)o3;
  if (out_f32) {
    float* of = out_f32 + (size_t)row * HID;
    of[tid] = o0; of[tid + 256] = o1; of[tid + 512] = o2; of[tid + 768] = o3;
  }
}

// ---------------------------------------------------------------------------
// Kernel 2: f32 -> bf16 convert
// ---------------------------------------------------------------------------
__launch_bounds__(256)
__global__ void cvt_bf16_kernel(const float* __restrict__ in, __bf16* __restrict__ out, int n) {
  int i = blockIdx.x * 256 + threadIdx.x;
  if (i < n) out[i] = (__bf16)in[i];
}

// ---------------------------------------------------------------------------
// WMMA fragment helpers (per ISA 7.12.2 layouts, wave32)
// ---------------------------------------------------------------------------
#define LDSK 40  // padded row stride in bf16 elements (80B -> conflict-free)

__device__ __forceinline__ v16bf load_afrag(const __bf16* tile, int rowbase, int l16, int lh) {
  // A 16x32 (MxK): lanes 0-15 hold K 0..7 & 16..23; lanes 16-31 hold K 8..15 & 24..31
  const __bf16* rp = tile + (size_t)(rowbase + l16) * LDSK;
  v8bf lo = *(const v8bf*)(rp + lh * 8);
  v8bf hi = *(const v8bf*)(rp + 16 + lh * 8);
  return __builtin_shufflevector(lo, hi, 0, 1, 2, 3, 4, 5, 6, 7, 8, 9, 10, 11, 12, 13, 14, 15);
}

__device__ __forceinline__ v16bf load_bfrag(const __bf16* tile, int rowbase, int l16, int lh) {
  // B 32x16 (KxN): lane holds one column; lanes 0-15 K 0..7 & 16..23; lanes 16-31 K 8..15 & 24..31
  const __bf16* rp = tile + (size_t)(rowbase + l16) * LDSK;
  v8bf lo = *(const v8bf*)(rp + lh * 8);
  v8bf hi = *(const v8bf*)(rp + 16 + lh * 8);
  return __builtin_shufflevector(lo, hi, 0, 1, 2, 3, 4, 5, 6, 7, 8, 9, 10, 11, 12, 13, 14, 15);
}

#define WMMA_BF16(a, b, c) \
  __builtin_amdgcn_wmma_f32_16x16x32_bf16(false, (a), false, (b), (short)0, (c), false, false)

// ---------------------------------------------------------------------------
// Async global->LDS copy of one 32-byte chunk (two b128 ops), ASYNCcnt-tracked
// ---------------------------------------------------------------------------
__device__ __forceinline__ void async_copy32(__bf16* lds, const __bf16* g) {
  const unsigned la = (unsigned)(unsigned long long)lds;
  asm volatile("global_load_async_to_lds_b128 %0, %1, off"
               :: "v"(la), "v"(g) : "memory");
  asm volatile("global_load_async_to_lds_b128 %0, %1, off offset:16"
               :: "v"(la), "v"(g) : "memory");
}

// Issue this thread's share of one 128x32 A tile + 128x32 B tile (4 async ops)
__device__ __forceinline__ void issue_tile(const __bf16* Ag, const __bf16* Bg, int K,
                                           int k0, __bf16* At, __bf16* Bt, int fr, int fc) {
  async_copy32(At + fr * LDSK + fc, Ag + (size_t)fr * K + k0 + fc);
  async_copy32(Bt + fr * LDSK + fc, Bg + (size_t)fr * K + k0 + fc);
}

// One 128x32 * 32x128 WMMA stage from a statically-named LDS buffer pair
__device__ __forceinline__ void stage_compute(const __bf16* At, const __bf16* Bt,
                                              v8f acc[2][4],
                                              int l16, int lh, int mrow, int ncol) {
  v16bf afr[2], bfr[4];
#pragma unroll
  for (int mi = 0; mi < 2; ++mi) afr[mi] = load_afrag(At, mrow * 32 + mi * 16, l16, lh);
#pragma unroll
  for (int ni = 0; ni < 4; ++ni) bfr[ni] = load_bfrag(Bt, ncol * 64 + ni * 16, l16, lh);
#pragma unroll
  for (int mi = 0; mi < 2; ++mi)
#pragma unroll
    for (int ni = 0; ni < 4; ++ni) acc[mi][ni] = WMMA_BF16(afr[mi], bfr[ni], acc[mi][ni]);
}

// ---------------------------------------------------------------------------
// Shared inner loop: acc += A_tile(128xK) * B_tile(128xK)^T, double-buffered
// async LDS tiles with STATIC buffer selection (keeps ds_load addrspace).
// Requires K % 64 == 0. Ag/Bg already offset to the 128-row tile starts.
// ---------------------------------------------------------------------------
__device__ __forceinline__ void wmma_tile_mm(const __bf16* __restrict__ Ag,
                                             const __bf16* __restrict__ Bg,
                                             int K,
                                             __bf16* At0, __bf16* At1,
                                             __bf16* Bt0, __bf16* Bt1,
                                             v8f acc[2][4], int tid) {
  const int lane = tid & 31, l16 = lane & 15, lh = lane >> 4;
  const int wave = tid >> 5, mrow = wave & 3, ncol = wave >> 2;
  const int fr = tid >> 1, fc = (tid & 1) * 16;

  issue_tile(Ag, Bg, K, 0, At0, Bt0, fr, fc);  // prime pipeline

  for (int k0 = 0; k0 < K; k0 += 64) {
    // ---- stage 0: consume buf0 (k0), prefetch buf1 (k0+32) ----
    issue_tile(Ag, Bg, K, k0 + 32, At1, Bt1, fr, fc);
    asm volatile("s_wait_asynccnt 0x4" ::: "memory");  // buf0's 4 older ops done
    __syncthreads();
    stage_compute(At0, Bt0, acc, l16, lh, mrow, ncol);
    __syncthreads();  // buf0 free for refill

    // ---- stage 1: consume buf1 (k0+32), prefetch buf0 (k0+64) ----
    if (k0 + 64 < K) {
      issue_tile(Ag, Bg, K, k0 + 64, At0, Bt0, fr, fc);
      asm volatile("s_wait_asynccnt 0x4" ::: "memory");
    } else {
      asm volatile("s_wait_asynccnt 0x0" ::: "memory");
    }
    __syncthreads();
    stage_compute(At1, Bt1, acc, l16, lh, mrow, ncol);
    __syncthreads();  // buf1 free for refill
  }
}

// ---------------------------------------------------------------------------
// Kernel 3: C[M,N](bf16) = A[M,K](bf16) * B[N,K]^T(bf16)   (NT, K contiguous)
// block = 256 thr (8 waves), tile 128x128, K step 32, double-buffered async
// ---------------------------------------------------------------------------
__launch_bounds__(256)
__global__ void wmma_gemm_nt_kernel(const __bf16* __restrict__ A,
                                    const __bf16* __restrict__ B,
                                    __bf16* __restrict__ C,
                                    int M, int N, int K) {
  __shared__ alignas(16) __bf16 At0[128 * LDSK], At1[128 * LDSK];
  __shared__ alignas(16) __bf16 Bt0[128 * LDSK], Bt1[128 * LDSK];
  const int tid = threadIdx.x;
  const int wave = tid >> 5, lane = tid & 31;
  const int l16 = lane & 15, lh = lane >> 4;
  const int mrow = wave & 3, ncol = wave >> 2;
  const int row0 = blockIdx.y * 128, col0 = blockIdx.x * 128;

  v8f acc[2][4];
#pragma unroll
  for (int mi = 0; mi < 2; ++mi)
#pragma unroll
    for (int ni = 0; ni < 4; ++ni) acc[mi][ni] = (v8f){};

  wmma_tile_mm(A + (size_t)row0 * K, B + (size_t)col0 * K, K,
               At0, At1, Bt0, Bt1, acc, tid);

  // C fragment layout: VGPR r -> row (lh*8 + r) of 16x16 tile, lane l16 -> column
#pragma unroll
  for (int mi = 0; mi < 2; ++mi) {
#pragma unroll
    for (int ni = 0; ni < 4; ++ni) {
      const int col = col0 + ncol * 64 + ni * 16 + l16;
      const int rowb = row0 + mrow * 32 + mi * 16 + lh * 8;
#pragma unroll
      for (int r = 0; r < 8; ++r)
        C[(size_t)(rowb + r) * N + col] = (__bf16)acc[mi][ni][r];
    }
  }
}

// ---------------------------------------------------------------------------
// Kernel 4: fused scores + max over S:
//   gmax[b, n] = scale * max_s sum_k Q[b,s,k] * Km[n,k]
// block handles one (b, 128-col tile); loops s over SEQ; no scores in memory.
// ---------------------------------------------------------------------------
__launch_bounds__(256)
__global__ void scores_max_kernel(const __bf16* __restrict__ Q,
                                  const __bf16* __restrict__ Km,
                                  float* __restrict__ gmax, float scale) {
  __shared__ alignas(16) __bf16 At0[128 * LDSK], At1[128 * LDSK];
  __shared__ alignas(16) __bf16 Bt0[128 * LDSK], Bt1[128 * LDSK];
  __shared__ float wred[8][64];
  const int tid = threadIdx.x;
  const int wave = tid >> 5, lane = tid & 31;
  const int l16 = lane & 15, lh = lane >> 4;
  const int b = blockIdx.y;
  const int col0 = blockIdx.x * 128;
  const __bf16* A = Q + (size_t)b * SEQ * HID;
  const __bf16* Bg = Km + (size_t)col0 * HID;

  float cmax[4] = {-INFINITY, -INFINITY, -INFINITY, -INFINITY};

  for (int s0 = 0; s0 < SEQ; s0 += 128) {
    v8f acc[2][4];
#pragma unroll
    for (int mi = 0; mi < 2; ++mi)
#pragma unroll
      for (int ni = 0; ni < 4; ++ni) acc[mi][ni] = (v8f){};

    wmma_tile_mm(A + (size_t)s0 * HID, Bg, HID,
                 At0, At1, Bt0, Bt1, acc, tid);

    // column-max: fold 8 rows per lane, then combine lane halves (rows 0-7 vs 8-15)
#pragma unroll
    for (int ni = 0; ni < 4; ++ni) {
      float m = -INFINITY;
#pragma unroll
      for (int mi = 0; mi < 2; ++mi) {
        v8f a = acc[mi][ni];
#pragma unroll
        for (int r = 0; r < 8; ++r) m = fmaxf(m, a[r]);
      }
      m = fmaxf(m, __shfl_xor(m, 16, 32));
      cmax[ni] = fmaxf(cmax[ni], m);
    }
  }

  if (lane < 16) {
#pragma unroll
    for (int ni = 0; ni < 4; ++ni) wred[wave][ni * 16 + l16] = cmax[ni];
  }
  __syncthreads();
  if (tid < 128) {
    const int nc = tid >> 6, jj = tid & 63;
    float m = wred[nc * 4 + 0][jj];
    m = fmaxf(m, wred[nc * 4 + 1][jj]);
    m = fmaxf(m, wred[nc * 4 + 2][jj]);
    m = fmaxf(m, wred[nc * 4 + 3][jj]);
    gmax[(size_t)b * NEMB + col0 + tid] = m * scale;
  }
}

// ---------------------------------------------------------------------------
// Kernel 5: top-64 cutoff + dense/sparse softmax blend per batch row
// ---------------------------------------------------------------------------
__launch_bounds__(256)
__global__ void topk_softmax_kernel(const float* __restrict__ gmax,
                                    const float* __restrict__ alpha_p,
                                    float* __restrict__ weights) {
  const int b = blockIdx.x;
  const int tid = threadIdx.x;
  __shared__ float vals[NEMB];
  __shared__ float wk[NEMB];
  __shared__ float red[256];
  __shared__ int redi[256];

  for (int j = tid; j < NEMB; j += 256) {
    float v = gmax[(size_t)b * NEMB + j];
    vals[j] = v;
    wk[j] = v;
  }
  __syncthreads();

  float gm = -INFINITY, cutoff = -INFINITY;
  for (int it = 0; it < KTOP; ++it) {
    float bv = -INFINITY; int bi = 0;
    for (int j = tid; j < NEMB; j += 256) {
      float v = wk[j];
      if (v > bv) { bv = v; bi = j; }
    }
    red[tid] = bv; redi[tid] = bi;
    __syncthreads();
    for (int s = 128; s > 0; s >>= 1) {
      if (tid < s && red[tid + s] > red[tid]) { red[tid] = red[tid + s]; redi[tid] = redi[tid + s]; }
      __syncthreads();
    }
    if (it == 0) gm = red[0];
    cutoff = red[0];
    if (tid == 0) wk[redi[0]] = -INFINITY;
    __syncthreads();
  }

  // dense & sparse sums (shared max gm)
  float sd = 0.0f, ss = 0.0f;
  for (int j = tid; j < NEMB; j += 256) {
    float e = __expf(vals[j] - gm);
    sd += e;
    if (vals[j] >= cutoff) ss += e;
  }
  red[tid] = sd; __syncthreads();
  for (int s = 128; s > 0; s >>= 1) { if (tid < s) red[tid] += red[tid + s]; __syncthreads(); }
  sd = red[0];
  __syncthreads();
  red[tid] = ss; __syncthreads();
  for (int s = 128; s > 0; s >>= 1) { if (tid < s) red[tid] += red[tid + s]; __syncthreads(); }
  ss = red[0];

  const float alpha = *alpha_p;
  const float rds = 1.0f / sd, rss = 1.0f / ss;
  for (int j = tid; j < NEMB; j += 256) {
    float e = __expf(vals[j] - gm);
    float dw = e * rds;
    float sw = (vals[j] >= cutoff) ? e * rss : 0.0f;
    weights[(size_t)b * NEMB + j] = alpha * dw + (1.0f - alpha) * sw;
  }
}

// ---------------------------------------------------------------------------
// Kernel 6: O = rmsnorm(weights @ En, on_w); skips zero-weight rows (sparse)
// ---------------------------------------------------------------------------
__launch_bounds__(256)
__global__ void out_proj_kernel(const float* __restrict__ weights,
                                const float* __restrict__ En,
                                const float* __restrict__ onw,
                                float* __restrict__ O) {
  const int b = blockIdx.x;
  const int tid = threadIdx.x;
  __shared__ float wsh[256];
  float a0 = 0, a1 = 0, a2 = 0, a3 = 0;
  for (int n0 = 0; n0 < NEMB; n0 += 256) {
    wsh[tid] = weights[(size_t)b * NEMB + n0 + tid];
    __syncthreads();
    for (int i = 0; i < 256; ++i) {
      float w = wsh[i];
      if (w != 0.0f) {
        const float* er = En + (size_t)(n0 + i) * HID;
        a0 += w * er[tid];       a1 += w * er[tid + 256];
        a2 += w * er[tid + 512]; a3 += w * er[tid + 768];
      }
    }
    __syncthreads();
  }
  __shared__ float red[256];
  red[tid] = a0 * a0 + a1 * a1 + a2 * a2 + a3 * a3;
  __syncthreads();
  for (int s = 128; s > 0; s >>= 1) { if (tid < s) red[tid] += red[tid + s]; __syncthreads(); }
  const float r = rsqrtf(red[0] * (1.0f / HID) + EPSV);
  float* o = O + (size_t)b * HID;
  o[tid] = a0 * r * onw[tid];
  o[tid + 256] = a1 * r * onw[tid + 256];
  o[tid + 512] = a2 * r * onw[tid + 512];
  o[tid + 768] = a3 * r * onw[tid + 768];
}

// ---------------------------------------------------------------------------
extern "C" void kernel_launch(void* const* d_in, const int* in_sizes, int n_in,
                              void* d_out, int out_size, void* d_ws, size_t ws_size,
                              hipStream_t stream) {
  const float* X     = (const float*)d_in[0];
  const float* alpha = (const float*)d_in[1];
  const float* E     = (const float*)d_in[2];
  const float* Wq    = (const float*)d_in[3];
  const float* Wk    = (const float*)d_in[4];
  const float* xn_w  = (const float*)d_in[5];
  const float* sn_w  = (const float*)d_in[6];
  const float* on_w  = (const float*)d_in[7];

  float* O = (float*)d_out;                   // 16 x 1024
  float* weights = O + (size_t)BATCH * HID;   // 16 x 4096

  // workspace carve (256B aligned chunks)
  char* p = (char*)d_ws;
  auto take = [&](size_t bytes) { char* q = p; p += (bytes + 255) & ~(size_t)255; return q; };
  __bf16* Xn  = (__bf16*)take((size_t)TOK * HID * 2);
  __bf16* Enb = (__bf16*)take((size_t)NEMB * HID * 2);
  float*  Enf = (float*) take((size_t)NEMB * HID * 4);
  __bf16* Wqb = (__bf16*)take((size_t)HID * HID * 2);
  __bf16* Wkb = (__bf16*)take((size_t)HID * HID * 2);
  __bf16* Qb  = (__bf16*)take((size_t)TOK * HID * 2);
  __bf16* Kmb = (__bf16*)take((size_t)NEMB * HID * 2);
  float*  gmx = (float*) take((size_t)BATCH * NEMB * 4);

  // 1) rmsnorm rows -> bf16 (E also kept in f32 for the output projection)
  rmsnorm_rows_kernel<<<TOK, 256, 0, stream>>>(X, xn_w, Xn, nullptr);
  rmsnorm_rows_kernel<<<NEMB, 256, 0, stream>>>(E, sn_w, Enb, Enf);

  // 2) weight conversion
  cvt_bf16_kernel<<<(HID * HID) / 256, 256, 0, stream>>>(Wq, Wqb, HID * HID);
  cvt_bf16_kernel<<<(HID * HID) / 256, 256, 0, stream>>>(Wk, Wkb, HID * HID);

  // 3) Q = Xn @ Wq^T ; Km = En @ Wk^T   (NT GEMMs, bf16 WMMA, async LDS tiles)
  wmma_gemm_nt_kernel<<<dim3(HID / 128, TOK / 128), 256, 0, stream>>>(Xn, Wqb, Qb, TOK, HID, HID);
  wmma_gemm_nt_kernel<<<dim3(HID / 128, NEMB / 128), 256, 0, stream>>>(Enb, Wkb, Kmb, NEMB, HID, HID);

  // 4) fused scores + max over S  (scale = HIDDEN^-0.5 = 1/32)
  scores_max_kernel<<<dim3(NEMB / 128, BATCH), 256, 0, stream>>>(Qb, Kmb, gmx, 1.0f / 32.0f);

  // 5) top-64 cutoff + softmax blend -> weights (second output)
  topk_softmax_kernel<<<BATCH, 256, 0, stream>>>(gmx, alpha, weights);

  // 6) O = rmsnorm(weights @ En, on_w)  (first output)
  out_proj_kernel<<<BATCH, 256, 0, stream>>>(weights, Enf, on_w, O);
}
